// DCGRUCell_22436909154350
// MI455X (gfx1250) — matrix operand order repaired and checked by reading, not compiled, not run on hardware
//
#include <hip/hip_runtime.h>
#include <hip/hip_bf16.h>

// ---------------------------------------------------------------------------
// DCGRU cell on MI455X (gfx1250, wave32, WMMA).
//
// Dominant cost: 4x (8192x8192 @ 8192x~66) diffusion matmuls streaming the
// normalized adjacency. Strategy: one-time normalize+transpose of adj into
// bf16 "Ant" (halves HBM bytes -> ~23us of A-traffic at 23.3 TB/s), then
// v_wmma_f32_16x16x32_bf16 with f32 accumulators. RHS is kept transposed
// (Vt[f][j], 1.3 MB, L2-resident) so BOTH WMMA operands load as contiguous
// 16B chunks per lane, and the D-layout lets us emit the next hop's
// transposed bf16 RHS as a single contiguous b128 store per lane.
// ---------------------------------------------------------------------------

#define NN    8192
#define DIN   2
#define UNITS 64
#define FEAT  66           // DIN + UNITS
#define FP    80           // FEAT padded to 5 x 16 WMMA column tiles
#define OUT_RU 128
#define OUT_C  64

typedef __bf16 bf16_t;
typedef __attribute__((ext_vector_type(16))) __bf16 v16bf;
typedef __attribute__((ext_vector_type(8)))  float  v8f;

union ABPack { uint4 u[2]; v16bf v; };
union TPack  { uint4 u; bf16_t h[8]; };

// ---------------- 1) row sums of (adj + I) -> dinv[j] = 1/rowsum ------------
__global__ __launch_bounds__(256)
void k_rowsum_inv(const float* __restrict__ adj, float* __restrict__ dinv) {
    const int j = blockIdx.x;
    float s = 0.0f;
    const float* row = adj + (size_t)j * NN;
    for (int i = threadIdx.x; i < NN; i += 256) s += row[i];
    __shared__ float red[256];
    red[threadIdx.x] = s;
    __syncthreads();
    for (int o = 128; o > 0; o >>= 1) {
        if (threadIdx.x < o) red[threadIdx.x] += red[threadIdx.x + o];
        __syncthreads();
    }
    if (threadIdx.x == 0) dinv[j] = 1.0f / (red[0] + 1.0f);  // +1 from I diag
}

// -------- 2) Ant[i][j] = bf16( (adj[j][i] + (i==j)) * dinv[j] ) -------------
// 32x32 LDS tile transpose: coalesced read of adj rows, coalesced write of Ant.
__global__ __launch_bounds__(256)
void k_norm_transpose(const float* __restrict__ adj,
                      const float* __restrict__ dinv,
                      bf16_t* __restrict__ Ant) {
    __shared__ float tile[32][33];
    const int i0 = blockIdx.x * 32;
    const int j0 = blockIdx.y * 32;
    const int tx = threadIdx.x & 31;
    const int ty = threadIdx.x >> 5;           // 0..7
#pragma unroll
    for (int t = 0; t < 4; ++t) {
        const int j = j0 + ty + 8 * t;
        const int i = i0 + tx;
        float v = adj[(size_t)j * NN + i];
        if (i == j) v += 1.0f;
        tile[ty + 8 * t][tx] = v * dinv[j];
    }
    __syncthreads();
#pragma unroll
    for (int t = 0; t < 4; ++t) {
        const int i = i0 + ty + 8 * t;
        const int j = j0 + tx;
        Ant[(size_t)i * NN + j] = (bf16_t)tile[tx][ty + 8 * t];
    }
}

// -------- 3) build x0 = concat(inp, state) padded to FP, + transposed bf16 --
// state = hx, optionally gated by r (gates[i*128 + 0..63], already sigmoided).
__global__ __launch_bounds__(256)
void k_build_x(const float* __restrict__ inp, const float* __restrict__ hx,
               const float* __restrict__ rgate,
               float* __restrict__ xf, bf16_t* __restrict__ xt) {
    const int i = blockIdx.x * blockDim.x + threadIdx.x;   // node id
    if (i >= NN) return;
#pragma unroll
    for (int f = 0; f < FP; ++f) {
        float v;
        if (f < DIN) {
            v = inp[i * DIN + f];
        } else if (f < FEAT) {
            const int u = f - DIN;
            float s = hx[i * UNITS + u];
            if (rgate) s *= rgate[i * OUT_RU + u];          // r * hx
            v = s;
        } else {
            v = 0.0f;                                       // pad columns
        }
        xf[(size_t)i * FP + f] = v;
        xt[(size_t)f * NN + i] = (bf16_t)v;                 // coalesced over i
    }
}

// -------- 4) diffusion matmul: out = alpha * Ant @ V + beta * addF ----------
// One wave per 16-row output strip x full 80 cols (5 WMMA tiles).
// A operand (16x32 bf16): lane m=lane&15, K-halves per ISA layout -> 2x b128.
// B operand (32x16 bf16): lane n=lane&15 reads Vt[f0+n][k0+bk+0..7] and [k0+bk+16..23] -> 2x b128.
// D (16x16 f32): lane holds col n, rows mb..mb+7 -> transposed bf16 out is one
// contiguous b128 store per lane.
__global__ __launch_bounds__(128)
void k_diffuse(const bf16_t* __restrict__ Ant, const bf16_t* __restrict__ Vt,
               const float* __restrict__ addF, float* __restrict__ outF,
               bf16_t* __restrict__ outT, float alpha, float beta) {
    const int lane  = threadIdx.x & 31;
    const int wave  = threadIdx.x >> 5;
    const int strip = blockIdx.x * 4 + wave;    // 0..511
    const int i0    = strip * 16;

    const int am = lane & 15;
    const int ak = (lane < 16) ? 0 : 8;
    const int bn = lane & 15;
    const int bk = (lane < 16) ? 0 : 8;

    v8f acc[5] = {};
    const bf16_t* aBase = Ant + (size_t)(i0 + am) * NN + ak;

    for (int k0 = 0; k0 < NN; k0 += 32) {
        ABPack A;
        A.u[0] = *(const uint4*)(aBase + k0);        // K = ak+0..7
        A.u[1] = *(const uint4*)(aBase + k0 + 16);   // K = ak+16..23
#pragma unroll
        for (int t = 0; t < 5; ++t) {
            ABPack B;
            const bf16_t* bp = Vt + (size_t)(t * 16 + bn) * NN + k0 + bk;
            B.u[0] = *(const uint4*)(bp);            // K = bk+0..7
            B.u[1] = *(const uint4*)(bp + 16);       // K = bk+16..23
            acc[t] = __builtin_amdgcn_wmma_f32_16x16x32_bf16(
                false, A.v, false, B.v, (short)0, acc[t], false, false);
        }
    }

    const int mb = (lane < 16) ? 0 : 8;
#pragma unroll
    for (int t = 0; t < 5; ++t) {
        const int fcol = t * 16 + bn;
        TPack tp;
#pragma unroll
        for (int r = 0; r < 8; ++r) {
            float v = alpha * acc[t][r];
            if (beta != 0.0f)
                v += beta * addF[(size_t)(i0 + mb + r) * FP + fcol];
            outF[(size_t)(i0 + mb + r) * FP + fcol] = v;
            tp.h[r] = (bf16_t)v;
        }
        if (outT)
            *(uint4*)(outT + (size_t)fcol * NN + i0 + mb) = tp.u;   // b128
    }
}

// -------- 5) gate projection: sigmoid(xk @ W_ru + b_ru) -> gates[N][128] ----
// xk feature index = f*3 + m (F-major, M inner per reference reshape).
__global__ __launch_bounds__(256)
void k_proj_gates(const float* __restrict__ x0, const float* __restrict__ x1,
                  const float* __restrict__ x2, const float* __restrict__ W,
                  const float* __restrict__ bias, float* __restrict__ gates) {
    const int idx = blockIdx.x * blockDim.x + threadIdx.x;   // N*128
    const int i = idx >> 7;
    const int o = idx & 127;
    const float* xr0 = x0 + (size_t)i * FP;
    const float* xr1 = x1 + (size_t)i * FP;
    const float* xr2 = x2 + (size_t)i * FP;
    float acc = bias[o];
    for (int f = 0; f < FEAT; ++f) {
        const float* wr = W + (size_t)(f * 3) * OUT_RU + o;
        acc = fmaf(xr0[f], wr[0 * OUT_RU], acc);
        acc = fmaf(xr1[f], wr[1 * OUT_RU], acc);
        acc = fmaf(xr2[f], wr[2 * OUT_RU], acc);
    }
    gates[idx] = 1.0f / (1.0f + __expf(-acc));
}

// -------- 6) candidate projection + GRU combine -----------------------------
__global__ __launch_bounds__(256)
void k_proj_c_final(const float* __restrict__ x0, const float* __restrict__ x1,
                    const float* __restrict__ x2, const float* __restrict__ W,
                    const float* __restrict__ bias,
                    const float* __restrict__ gates,
                    const float* __restrict__ hx, float* __restrict__ out) {
    const int idx = blockIdx.x * blockDim.x + threadIdx.x;   // N*64
    const int i = idx >> 6;
    const int o = idx & 63;
    const float* xr0 = x0 + (size_t)i * FP;
    const float* xr1 = x1 + (size_t)i * FP;
    const float* xr2 = x2 + (size_t)i * FP;
    float acc = bias[o];
    for (int f = 0; f < FEAT; ++f) {
        const float* wr = W + (size_t)(f * 3) * OUT_C + o;
        acc = fmaf(xr0[f], wr[0 * OUT_C], acc);
        acc = fmaf(xr1[f], wr[1 * OUT_C], acc);
        acc = fmaf(xr2[f], wr[2 * OUT_C], acc);
    }
    const float c = tanhf(acc);
    const float u = gates[i * OUT_RU + UNITS + o];
    out[idx] = u * hx[idx] + (1.0f - u) * c;
}

// ---------------------------------------------------------------------------
extern "C" void kernel_launch(void* const* d_in, const int* in_sizes, int n_in,
                              void* d_out, int out_size, void* d_ws, size_t ws_size,
                              hipStream_t stream) {
    const float* inputs = (const float*)d_in[0];   // [N*2]
    const float* hx     = (const float*)d_in[1];   // [N*64]
    const float* adj    = (const float*)d_in[2];   // [N*N]
    const float* w_ru   = (const float*)d_in[3];   // [198*128]
    const float* b_ru   = (const float*)d_in[4];   // [128]
    const float* w_c    = (const float*)d_in[5];   // [198*64]
    const float* b_c    = (const float*)d_in[6];   // [64]
    float* out = (float*)d_out;                    // [N*64]

    // workspace layout
    char* ws = (char*)d_ws;
    size_t off = 0;
    bf16_t* Ant = (bf16_t*)(ws + off); off += (size_t)NN * NN * sizeof(bf16_t);
    float*  dinv = (float*)(ws + off); off += (size_t)NN * sizeof(float);
    float*  x0f  = (float*)(ws + off); off += (size_t)NN * FP * sizeof(float);
    float*  x1f  = (float*)(ws + off); off += (size_t)NN * FP * sizeof(float);
    float*  x2f  = (float*)(ws + off); off += (size_t)NN * FP * sizeof(float);
    bf16_t* x0t  = (bf16_t*)(ws + off); off += (size_t)FP * NN * sizeof(bf16_t);
    bf16_t* x1t  = (bf16_t*)(ws + off); off += (size_t)FP * NN * sizeof(bf16_t);
    float*  gates = (float*)(ws + off); off += (size_t)NN * OUT_RU * sizeof(float);
    (void)ws_size; (void)n_in; (void)in_sizes; (void)out_size;

    // 1-2) normalized, transposed bf16 adjacency (one-time, amortized 4x)
    k_rowsum_inv<<<NN, 256, 0, stream>>>(adj, dinv);
    k_norm_transpose<<<dim3(NN / 32, NN / 32), 256, 0, stream>>>(adj, dinv, Ant);

    const dim3 dgrid(128), dblk(128);   // 512 strips, 4 waves/block

    // ---- gconv #1: state = hx ----
    k_build_x<<<NN / 256, 256, 0, stream>>>(inputs, hx, nullptr, x0f, x0t);
    // x1 = Ant @ x0
    k_diffuse<<<dgrid, dblk, 0, stream>>>(Ant, x0t, nullptr, x1f, x1t, 1.0f, 0.0f);
    // x2 = 2*(Ant @ x1) - x0
    k_diffuse<<<dgrid, dblk, 0, stream>>>(Ant, x1t, x0f, x2f, nullptr, 2.0f, -1.0f);
    k_proj_gates<<<(NN * OUT_RU) / 256, 256, 0, stream>>>(x0f, x1f, x2f,
                                                          w_ru, b_ru, gates);

    // ---- gconv #2: state = r * hx ----
    k_build_x<<<NN / 256, 256, 0, stream>>>(inputs, hx, gates, x0f, x0t);
    k_diffuse<<<dgrid, dblk, 0, stream>>>(Ant, x0t, nullptr, x1f, x1t, 1.0f, 0.0f);
    k_diffuse<<<dgrid, dblk, 0, stream>>>(Ant, x1t, x0f, x2f, nullptr, 2.0f, -1.0f);

    // candidate + GRU combine
    k_proj_c_final<<<(NN * OUT_C) / 256, 256, 0, stream>>>(x0f, x1f, x2f,
                                                           w_c, b_c, gates, hx, out);
}
